// NeuroHybridAttention_20753281974645
// MI455X (gfx1250) — compile-verified, matrix operand order checked by reading, not run on hardware
//
#include <hip/hip_runtime.h>
#include <hip/hip_bf16.h>
#include <stdint.h>
#include <stddef.h>

#define B_    2
#define S_    4096
#define H_    32
#define KVH_  8
#define D_    64
#define HID_  2048
#define WIN_  256
#define CHK_  128
#define BS_   (B_*S_)

typedef __attribute__((ext_vector_type(16))) __bf16        v16bf;
typedef __attribute__((ext_vector_type(8)))  float         v8f;
typedef __attribute__((ext_vector_type(8)))  unsigned short su8;

union FragU { su8 h[2]; v16bf v; };

__device__ __forceinline__ unsigned short f2bf(float f) {
  unsigned int u = __builtin_bit_cast(unsigned int, f);
  u += 0x7FFFu + ((u >> 16) & 1u);               // RNE truncate
  return (unsigned short)(u >> 16);
}
__device__ __forceinline__ float bf2f(unsigned short h) {
  unsigned int u = ((unsigned int)h) << 16;
  return __builtin_bit_cast(float, u);
}
// A/B fragment: elements [p..p+7] and [p+16..p+23] (per 16-bit WMMA VGPR layout)
__device__ __forceinline__ v16bf ldfrag(const unsigned short* p) {
  FragU f;
  f.h[0] = *(const su8*)p;
  f.h[1] = *(const su8*)(p + 16);
  return f.v;
}
#define WMMA_BF16(A, Bf, C) \
  __builtin_amdgcn_wmma_f32_16x16x32_bf16(false, (A), false, (Bf), (short)0, (C), false, false)

// Async global->LDS 16B/lane copy (GLOBAL_LOAD_ASYNC_TO_LDS_B128, ASYNCcnt).
// LDS dest = low 32 bits of generic pointer (== wave-relative LDS byte offset per ISA aperture rules).
#define ASYNC_CP_B128(lds_ptr, gl_ptr)                                        \
  asm volatile("global_load_async_to_lds_b128 %0, %1, off"                    \
               :: "v"((unsigned int)(size_t)(lds_ptr)), "v"(gl_ptr) : "memory")
#define ASYNC_WAIT0() asm volatile("s_wait_asynccnt 0x0" ::: "memory")

// ---------------------------------------------------------------- cast fp32 -> bf16
__global__ __launch_bounds__(256) void cast_f32_bf16(const float* __restrict__ in,
                                                     unsigned short* __restrict__ out, int n) {
  int i = blockIdx.x * 256 + threadIdx.x;
  if (i < n) out[i] = f2bf(in[i]);
}

// ---------------------------------------------------------------- generic bf16 WMMA GEMM
// C[M,N] (f32) = A[M,K] (bf16, row-major) x W[N,K]^T (bf16, row-major over K)
// Double-buffered LDS staging via async-to-LDS DMA.
__global__ __launch_bounds__(256) void gemm_bf16(const unsigned short* __restrict__ A,
                                                 const unsigned short* __restrict__ W,
                                                 float* __restrict__ C, int M, int N, int K) {
  __shared__ unsigned short As[2][128 * 32];
  __shared__ unsigned short Bs[2][128 * 32];
  const int t = threadIdx.x;
  const int lane = t & 31, wave = t >> 5;
  const int lrow = lane & 15;
  const int lkb  = (lane >> 4) << 3;   // K-half base (0 or 8)
  const int rb   = lkb;                // C-row base (0 or 8)
  const int m0 = blockIdx.y * 128;
  const int n0 = blockIdx.x * 128;
  const int wm = (wave & 1) * 64;      // 2 row-groups of 64
  const int wn = (wave >> 1) * 32;     // 4 col-groups of 32

  // per-thread staging coordinates: two b128 chunks per matrix per tile
  const int e0 = t << 3;
  const int e1 = (t + 256) << 3;
  const int r0 = e0 >> 5, c0 = e0 & 31;
  const int r1 = e1 >> 5, c1 = e1 & 31;
  const unsigned short* gA0 = &A[(size_t)(m0 + r0) * K + c0];
  const unsigned short* gA1 = &A[(size_t)(m0 + r1) * K + c1];
  const unsigned short* gB0 = &W[(size_t)(n0 + r0) * K + c0];
  const unsigned short* gB1 = &W[(size_t)(n0 + r1) * K + c1];

  const v8f vz = {};
  v8f acc[4][2];
#pragma unroll
  for (int i = 0; i < 4; ++i)
#pragma unroll
    for (int j = 0; j < 2; ++j) acc[i][j] = vz;

  // prologue: DMA first K-panel into buffer 0
  ASYNC_CP_B128(&As[0][r0 * 32 + c0], gA0);
  ASYNC_CP_B128(&As[0][r1 * 32 + c1], gA1);
  ASYNC_CP_B128(&Bs[0][r0 * 32 + c0], gB0);
  ASYNC_CP_B128(&Bs[0][r1 * 32 + c1], gB1);
  ASYNC_WAIT0();
  __syncthreads();

  int cur = 0;
  for (int k0 = 0; k0 < K; k0 += 32) {
    if (k0 + 32 < K) {               // prefetch next K-panel into the other buffer
      const int nxt = cur ^ 1;
      ASYNC_CP_B128(&As[nxt][r0 * 32 + c0], gA0 + k0 + 32);
      ASYNC_CP_B128(&As[nxt][r1 * 32 + c1], gA1 + k0 + 32);
      ASYNC_CP_B128(&Bs[nxt][r0 * 32 + c0], gB0 + k0 + 32);
      ASYNC_CP_B128(&Bs[nxt][r1 * 32 + c1], gB1 + k0 + 32);
    }
    v16bf bf[2];
#pragma unroll
    for (int nt = 0; nt < 2; ++nt)
      bf[nt] = ldfrag(&Bs[cur][(wn + nt * 16 + lrow) * 32 + lkb]);
#pragma unroll
    for (int mt = 0; mt < 4; ++mt) {
      v16bf af = ldfrag(&As[cur][(wm + mt * 16 + lrow) * 32 + lkb]);
#pragma unroll
      for (int nt = 0; nt < 2; ++nt) acc[mt][nt] = WMMA_BF16(af, bf[nt], acc[mt][nt]);
    }
    ASYNC_WAIT0();
    __syncthreads();
    cur ^= 1;
  }
#pragma unroll
  for (int mt = 0; mt < 4; ++mt)
#pragma unroll
    for (int nt = 0; nt < 2; ++nt) {
      int n = n0 + wn + nt * 16 + lrow;
      int mb = m0 + wm + mt * 16 + rb;
#pragma unroll
      for (int r = 0; r < 8; ++r) C[(size_t)(mb + r) * N + n] = acc[mt][nt][r];
    }
}

// ---------------------------------------------------------------- RoPE + head-major relayout
__global__ __launch_bounds__(256) void rope_reshape(const float* __restrict__ Qf,
                                                    const float* __restrict__ Kf,
                                                    const float* __restrict__ Vf,
                                                    const float* __restrict__ cosb,
                                                    const float* __restrict__ sinb,
                                                    unsigned short* __restrict__ Qr,
                                                    unsigned short* __restrict__ Kr,
                                                    unsigned short* __restrict__ Vr) {
  unsigned int idx = blockIdx.x * 256u + threadIdx.x;   // over B*S*H*D
  int d = idx & 63u;
  unsigned int r1 = idx >> 6;
  int h = r1 & 31u;
  unsigned int r2 = r1 >> 5;
  int s = r2 & 4095u;
  int b = r2 >> 12;
  size_t bs = (size_t)b * S_ + s;

  float c  = cosb[bs * 64 + d];
  float sn = sinb[bs * 64 + d];

  float q = Qf[bs * HID_ + h * 64 + d];
  float qrot = (d < 32) ? -Qf[bs * HID_ + h * 64 + d + 32] : Qf[bs * HID_ + h * 64 + d - 32];
  Qr[((size_t)(b * H_ + h) * S_ + s) * 64 + d] = f2bf(q * c + qrot * sn);

  if (h < KVH_) {
    size_t kb = bs * (KVH_ * D_) + h * 64 + d;
    float k = Kf[kb];
    float krot = (d < 32) ? -Kf[kb + 32] : Kf[kb - 32];
    Kr[((size_t)(b * KVH_ + h) * S_ + s) * 64 + d] = f2bf(k * c + krot * sn);
    Vr[((size_t)(b * KVH_ + h) * S_ + s) * 64 + d] = f2bf(Vf[kb]);
  }
}

// ---------------------------------------------------------------- local window attention (flash-style)
__global__ __launch_bounds__(256) void local_attn(const unsigned short* __restrict__ Qr,
                                                  const unsigned short* __restrict__ Kr,
                                                  const unsigned short* __restrict__ Vr,
                                                  float* __restrict__ Lout) {
  __shared__ unsigned short Qs[256 * 64];   // 32 KB
  __shared__ unsigned short Ks[32 * 64];    //  4 KB  (per 32-key chunk)
  __shared__ unsigned short Vts[64 * 32];   //  4 KB  (transposed: [d][key])
  __shared__ unsigned short Ps[8 * 32 * 32];// 16 KB  (per-wave prob tiles)

  const int nb = blockIdx.x, h = blockIdx.y, b = blockIdx.z;
  const int kvh = h >> 2;
  const int t = threadIdx.x;
  const int lane = t & 31, wave = t >> 5;
  const int lrow = lane & 15, lkb = (lane >> 4) << 3, rb = lkb;
  const int wq0 = wave * 32;

  const size_t qbase = ((size_t)(b * H_ + h) * S_ + (size_t)nb * WIN_) * 64;
  const size_t kbase = (size_t)(b * KVH_ + kvh) * S_ * 64;

  // async DMA the whole 256x64 Q tile into LDS (full EXEC, always in-bounds)
#pragma unroll
  for (int p = 0; p < 8; ++p) {
    int e = (t + p * 256) << 3;
    ASYNC_CP_B128(&Qs[e], &Qr[qbase + e]);
  }
  ASYNC_WAIT0();

  const v8f vz = {};
  v8f acc_o[2][4];
  float m_run[2][8], l_run[2][8];
#pragma unroll
  for (int mt = 0; mt < 2; ++mt) {
#pragma unroll
    for (int nt = 0; nt < 4; ++nt) acc_o[mt][nt] = vz;
#pragma unroll
    for (int r = 0; r < 8; ++r) { m_run[mt][r] = -1e30f; l_run[mt][r] = 0.f; }
  }

  for (int kc = 0; kc < 16; ++kc) {
    const int j0 = kc * 32;
    __syncthreads();
    {   // stage 32 keys (K row-major, V transposed) -- predicated zero-fill, keep sync
      int e = t << 3;
      int kk = e >> 6, d0 = e & 63;
      int keyg = nb * WIN_ + j0 + kk - WIN_;
      su8 kv8 = {}, vv8 = {};
      if (keyg >= 0) {
        kv8 = *(const su8*)&Kr[kbase + (size_t)keyg * 64 + d0];
        vv8 = *(const su8*)&Vr[kbase + (size_t)keyg * 64 + d0];
      }
      *(su8*)&Ks[kk * 64 + d0] = kv8;
#pragma unroll
      for (int i = 0; i < 8; ++i) Vts[(d0 + i) * 32 + kk] = vv8[i];
    }
    __syncthreads();

    v8f sacc[2][2];
#pragma unroll
    for (int mt = 0; mt < 2; ++mt)
#pragma unroll
      for (int nt = 0; nt < 2; ++nt) sacc[mt][nt] = vz;
#pragma unroll
    for (int ks = 0; ks < 2; ++ks) {
      v16bf bf0 = ldfrag(&Ks[(lrow) * 64 + ks * 32 + lkb]);
      v16bf bf1 = ldfrag(&Ks[(16 + lrow) * 64 + ks * 32 + lkb]);
#pragma unroll
      for (int mt = 0; mt < 2; ++mt) {
        v16bf af = ldfrag(&Qs[(wq0 + mt * 16 + lrow) * 64 + ks * 32 + lkb]);
        sacc[mt][0] = WMMA_BF16(af, bf0, sacc[mt][0]);
        sacc[mt][1] = WMMA_BF16(af, bf1, sacc[mt][1]);
      }
    }
    // mask + online softmax + stage P (bf16) for the PV matmul
#pragma unroll
    for (int mt = 0; mt < 2; ++mt) {
#pragma unroll
      for (int r = 0; r < 8; ++r) {
        int qi = wq0 + mt * 16 + rb + r;        // in-window query index
        float v0, v1;
        {
          int j = j0 + lrow;
          int diff = qi + WIN_ - j;
          int keyg = nb * WIN_ + j - WIN_;
          v0 = (diff >= 0 && diff < WIN_ && keyg >= 0) ? sacc[mt][0][r] * 0.125f : -1e30f;
        }
        {
          int j = j0 + 16 + lrow;
          int diff = qi + WIN_ - j;
          int keyg = nb * WIN_ + j - WIN_;
          v1 = (diff >= 0 && diff < WIN_ && keyg >= 0) ? sacc[mt][1][r] * 0.125f : -1e30f;
        }
        float rm = fmaxf(v0, v1);
#pragma unroll
        for (int o = 1; o < 16; o <<= 1) rm = fmaxf(rm, __shfl_xor(rm, o, 32));
        float mold = m_run[mt][r];
        float mnew = fmaxf(mold, rm);
        float p0 = (v0 > -1e29f) ? __expf(v0 - mnew) : 0.f;
        float p1 = (v1 > -1e29f) ? __expf(v1 - mnew) : 0.f;
        float rs = p0 + p1;
#pragma unroll
        for (int o = 1; o < 16; o <<= 1) rs += __shfl_xor(rs, o, 32);
        float alpha = (mold > -1e29f) ? __expf(mold - mnew) : 0.f;
        l_run[mt][r] = l_run[mt][r] * alpha + rs;
        m_run[mt][r] = mnew;
#pragma unroll
        for (int nt2 = 0; nt2 < 4; ++nt2) acc_o[mt][nt2][r] *= alpha;
        int prow = mt * 16 + rb + r;            // 0..31 inside this wave's P tile
        Ps[wave * 1024 + prow * 32 + lrow]      = f2bf(p0);
        Ps[wave * 1024 + prow * 32 + 16 + lrow] = f2bf(p1);
      }
    }
    // PV: P(32x32) x V(32x64)
#pragma unroll
    for (int mt = 0; mt < 2; ++mt) {
      v16bf pf = ldfrag(&Ps[wave * 1024 + (mt * 16 + lrow) * 32 + lkb]);
#pragma unroll
      for (int nt2 = 0; nt2 < 4; ++nt2) {
        v16bf vf = ldfrag(&Vts[(nt2 * 16 + lrow) * 32 + lkb]);
        acc_o[mt][nt2] = WMMA_BF16(pf, vf, acc_o[mt][nt2]);
      }
    }
  }
#pragma unroll
  for (int mt = 0; mt < 2; ++mt)
#pragma unroll
    for (int nt2 = 0; nt2 < 4; ++nt2)
#pragma unroll
      for (int r = 0; r < 8; ++r) {
        int qi = wq0 + mt * 16 + rb + r;
        int d = nt2 * 16 + lrow;
        Lout[qbase + (size_t)qi * 64 + d] = acc_o[mt][nt2][r] / l_run[mt][r];
      }
}

// ---------------------------------------------------------------- chunked linear attention + combine
__global__ __launch_bounds__(256) void linear_attn(const unsigned short* __restrict__ Qr,
                                                   const unsigned short* __restrict__ Kr,
                                                   const unsigned short* __restrict__ Vr,
                                                   const float* __restrict__ Lout,
                                                   unsigned short* __restrict__ Hyb) {
  __shared__ unsigned short Pq[128 * 64];    // 16 KB  phi(q) row-major
  __shared__ unsigned short Pk[128 * 64];    // 16 KB  phi(k) row-major
  __shared__ unsigned short Pkt[64 * 128];   // 16 KB  phi(k) transposed [d][k]
  __shared__ unsigned short Vt[64 * 128];    // 16 KB  v transposed [e][k]
  __shared__ unsigned short Pls[128 * 128];  // 32 KB  masked scores (bf16)
  __shared__ unsigned short Sbf[64 * 64];    //  8 KB  S_prev bf16, layout [e][d]
  __shared__ float Sst[64 * 64];             // 16 KB  running KV state, layout [e][d]
  __shared__ float zst[64];
  __shared__ float den[128];

  const int bh = blockIdx.x;
  const int b = bh >> 5, h = bh & 31;
  const int kvh = h >> 2;
  const int t = threadIdx.x;
  const int lane = t & 31, wave = t >> 5;
  const int lrow = lane & 15, lkb = (lane >> 4) << 3, rb = lkb;

  const size_t qhead = (size_t)(b * H_ + h) * S_ * 64;
  const size_t khead = (size_t)(b * KVH_ + kvh) * S_ * 64;

  for (int i = t; i < 64 * 64; i += 256) Sst[i] = 0.f;
  if (t < 64) zst[t] = 0.f;

  const v8f vz = {};
  for (int n = 0; n < 32; ++n) {
    const int s0 = n * CHK_;
    __syncthreads();
    // ---- stage phi(q), phi(k), v (+transposed copies), snapshot S_prev ----
#pragma unroll
    for (int p = 0; p < 4; ++p) {
      int e = (t + p * 256) << 3;
      int c = e >> 6, d0 = e & 63;
      su8 qv = *(const su8*)&Qr[qhead + (size_t)(s0 + c) * 64 + d0];
      su8 kv = *(const su8*)&Kr[khead + (size_t)(s0 + c) * 64 + d0];
      su8 vv = *(const su8*)&Vr[khead + (size_t)(s0 + c) * 64 + d0];
      su8 pq8, pk8;
#pragma unroll
      for (int i = 0; i < 8; ++i) {
        float q = bf2f(qv[i]); q = (q > 0.f) ? q + 1.f : __expf(q);   // elu(x)+1
        float k = bf2f(kv[i]); k = (k > 0.f) ? k + 1.f : __expf(k);
        pq8[i] = f2bf(q);
        unsigned short kb16 = f2bf(k);
        pk8[i] = kb16;
        Pkt[(d0 + i) * 128 + c] = kb16;
        Vt[(d0 + i) * 128 + c]  = vv[i];
      }
      *(su8*)&Pq[c * 64 + d0] = pq8;
      *(su8*)&Pk[c * 64 + d0] = pk8;
    }
    for (int i = t; i < 64 * 64; i += 256) Sbf[i] = f2bf(Sst[i]);
    if (t < 128) den[t] = 0.f;
    __syncthreads();

    // ---- intra-chunk scores (masked, no softmax) + den ----
    {
      const int rg = (wave & 3) * 32;
      const int cg = (wave >> 2) * 64;
      v8f sacc[2][4];
#pragma unroll
      for (int mt = 0; mt < 2; ++mt)
#pragma unroll
        for (int nt = 0; nt < 4; ++nt) sacc[mt][nt] = vz;
#pragma unroll
      for (int ks = 0; ks < 2; ++ks) {
        v16bf bfr[4];
#pragma unroll
        for (int nt = 0; nt < 4; ++nt)
          bfr[nt] = ldfrag(&Pk[(cg + nt * 16 + lrow) * 64 + ks * 32 + lkb]);
#pragma unroll
        for (int mt = 0; mt < 2; ++mt) {
          v16bf af = ldfrag(&Pq[(rg + mt * 16 + lrow) * 64 + ks * 32 + lkb]);
#pragma unroll
          for (int nt = 0; nt < 4; ++nt) sacc[mt][nt] = WMMA_BF16(af, bfr[nt], sacc[mt][nt]);
        }
      }
#pragma unroll
      for (int mt = 0; mt < 2; ++mt) {
#pragma unroll
        for (int r = 0; r < 8; ++r) {
          int crow = rg + mt * 16 + rb + r;
          float rsum = 0.f;
#pragma unroll
          for (int nt = 0; nt < 4; ++nt) {
            int kcol = cg + nt * 16 + lrow;
            float v = (kcol <= crow) ? sacc[mt][nt][r] : 0.f;   // tril mask
            Pls[crow * 128 + kcol] = f2bf(v);
            rsum += v;
          }
#pragma unroll
          for (int o = 1; o < 16; o <<= 1) rsum += __shfl_xor(rsum, o, 32);
          if (lrow == 0) atomicAdd(&den[crow], rsum);
        }
      }
      if (t < 128) {      // den += phi(q) . z_prev
        float s = 0.f;
        for (int d = 0; d < 64; ++d) s += bf2f(Pq[t * 64 + d]) * zst[d];
        atomicAdd(&den[t], s);
      }
    }
    __syncthreads();

    // ---- numerator (intra + inter) -> combine with local, write hybrid ----
    {
      const int rg = (wave & 3) * 32;
      const int cg = (wave >> 2) * 32;
      v8f oacc[2][2];
#pragma unroll
      for (int mt = 0; mt < 2; ++mt)
#pragma unroll
        for (int nt = 0; nt < 2; ++nt) oacc[mt][nt] = vz;
#pragma unroll
      for (int ks = 0; ks < 4; ++ks) {     // intra: scores(128K) x V
        v16bf bfr[2];
#pragma unroll
        for (int nt = 0; nt < 2; ++nt)
          bfr[nt] = ldfrag(&Vt[(cg + nt * 16 + lrow) * 128 + ks * 32 + lkb]);
#pragma unroll
        for (int mt = 0; mt < 2; ++mt) {
          v16bf af = ldfrag(&Pls[(rg + mt * 16 + lrow) * 128 + ks * 32 + lkb]);
#pragma unroll
          for (int nt = 0; nt < 2; ++nt) oacc[mt][nt] = WMMA_BF16(af, bfr[nt], oacc[mt][nt]);
        }
      }
#pragma unroll
      for (int ks = 0; ks < 2; ++ks) {     // inter: phi(q)(64K) x S_prev
        v16bf bfr[2];
#pragma unroll
        for (int nt = 0; nt < 2; ++nt)
          bfr[nt] = ldfrag(&Sbf[(cg + nt * 16 + lrow) * 64 + ks * 32 + lkb]);
#pragma unroll
        for (int mt = 0; mt < 2; ++mt) {
          v16bf af = ldfrag(&Pq[(rg + mt * 16 + lrow) * 64 + ks * 32 + lkb]);
#pragma unroll
          for (int nt = 0; nt < 2; ++nt) oacc[mt][nt] = WMMA_BF16(af, bfr[nt], oacc[mt][nt]);
        }
      }
#pragma unroll
      for (int mt = 0; mt < 2; ++mt)
#pragma unroll
        for (int nt = 0; nt < 2; ++nt)
#pragma unroll
          for (int r = 0; r < 8; ++r) {
            int crow = rg + mt * 16 + rb + r;
            int e = cg + nt * 16 + lrow;
            int sg = s0 + crow;
            float lin = oacc[mt][nt][r] / (den[crow] + 1e-6f);
            float loc = Lout[qhead + (size_t)sg * 64 + e];
            Hyb[((size_t)b * S_ + sg) * HID_ + h * 64 + e] = f2bf(0.5f * (loc + lin));
          }
    }

    // ---- state update: S += phi(k)^T v ; z += sum_k phi(k) ----
    {
      const int mt = wave & 3;            // d-tile
      const int ng = (wave >> 2) * 2;     // e-tiles ng, ng+1
#pragma unroll
      for (int j = 0; j < 2; ++j) {
        int ecol = (ng + j) * 16 + lrow;
        int dbase = mt * 16 + rb;
        v8f cacc = *(const v8f*)&Sst[ecol * 64 + dbase];
#pragma unroll
        for (int ks = 0; ks < 4; ++ks) {
          v16bf af  = ldfrag(&Pkt[(mt * 16 + lrow) * 128 + ks * 32 + lkb]);
          v16bf bfr = ldfrag(&Vt[((ng + j) * 16 + lrow) * 128 + ks * 32 + lkb]);
          cacc = WMMA_BF16(af, bfr, cacc);
        }
        *(v8f*)&Sst[ecol * 64 + dbase] = cacc;
      }
      if (t < 64) {
        float s = zst[t];
        for (int k = 0; k < 128; ++k) s += bf2f(Pkt[t * 128 + k]);
        zst[t] = s;
      }
    }
  }
}

// ---------------------------------------------------------------- host launcher
extern "C" void kernel_launch(void* const* d_in, const int* in_sizes, int n_in,
                              void* d_out, int out_size, void* d_ws, size_t ws_size,
                              hipStream_t stream) {
  (void)in_sizes; (void)n_in; (void)out_size; (void)ws_size;
  const float* hidden = (const float*)d_in[0];
  const float* cosb   = (const float*)d_in[1];
  const float* sinb   = (const float*)d_in[2];
  const float* Wq     = (const float*)d_in[3];
  const float* Wk     = (const float*)d_in[4];
  const float* Wv     = (const float*)d_in[5];
  const float* Wo     = (const float*)d_in[6];
  float* out = (float*)d_out;
  char* ws = (char*)d_ws;

  // workspace layout (bytes)
  const size_t oXb  = 0;                       // X  bf16           33.5 MB
  const size_t oWqb = oXb  + 33554432;         // Wq bf16            8.4 MB
  const size_t oWkb = oWqb + 8388608;          // Wk bf16            2.1 MB
  const size_t oWvb = oWkb + 2097152;          // Wv bf16            2.1 MB
  const size_t oWob = oWvb + 2097152;          // Wo bf16            8.4 MB
  const size_t oQf  = oWob + 8388608;          // Q f32 / Lout f32  67.1 MB (aliased)
  const size_t oKf  = oQf  + 67108864;         // K f32             16.8 MB
  const size_t oVf  = oKf  + 16777216;         // V f32             16.8 MB
  const size_t oQr  = oVf  + 16777216;         // Q roped bf16      33.5 MB
  const size_t oKr  = oQr  + 33554432;         // K roped bf16       8.4 MB
  const size_t oVr  = oKr  + 8388608;          // V bf16             8.4 MB
  const size_t oHyb = oVr  + 8388608;          // hybrid bf16       33.5 MB

  unsigned short* Xb  = (unsigned short*)(ws + oXb);
  unsigned short* Wqb = (unsigned short*)(ws + oWqb);
  unsigned short* Wkb = (unsigned short*)(ws + oWkb);
  unsigned short* Wvb = (unsigned short*)(ws + oWvb);
  unsigned short* Wob = (unsigned short*)(ws + oWob);
  float*          Qf  = (float*)(ws + oQf);
  float*          Kf  = (float*)(ws + oKf);
  float*          Vf  = (float*)(ws + oVf);
  unsigned short* Qr  = (unsigned short*)(ws + oQr);
  unsigned short* Kr  = (unsigned short*)(ws + oKr);
  unsigned short* Vr  = (unsigned short*)(ws + oVr);
  float*          Lout = (float*)(ws + oQf);    // alias: Qf dead after rope
  unsigned short* Hyb = (unsigned short*)(ws + oHyb);

  // 1) casts
  cast_f32_bf16<<<(BS_*HID_ + 255) / 256, 256, 0, stream>>>(hidden, Xb, BS_*HID_);
  cast_f32_bf16<<<(HID_*HID_ + 255) / 256, 256, 0, stream>>>(Wq, Wqb, HID_*HID_);
  cast_f32_bf16<<<(KVH_*D_*HID_ + 255) / 256, 256, 0, stream>>>(Wk, Wkb, KVH_*D_*HID_);
  cast_f32_bf16<<<(KVH_*D_*HID_ + 255) / 256, 256, 0, stream>>>(Wv, Wvb, KVH_*D_*HID_);
  cast_f32_bf16<<<(HID_*HID_ + 255) / 256, 256, 0, stream>>>(Wo, Wob, HID_*HID_);

  // 2) projections (WMMA GEMM, async double-buffered)
  gemm_bf16<<<dim3(HID_/128, BS_/128), 256, 0, stream>>>(Xb, Wqb, Qf, BS_, HID_, HID_);
  gemm_bf16<<<dim3((KVH_*D_)/128, BS_/128), 256, 0, stream>>>(Xb, Wkb, Kf, BS_, KVH_*D_, HID_);
  gemm_bf16<<<dim3((KVH_*D_)/128, BS_/128), 256, 0, stream>>>(Xb, Wvb, Vf, BS_, KVH_*D_, HID_);

  // 3) RoPE + relayout
  rope_reshape<<<(B_*S_*H_*D_) / 256, 256, 0, stream>>>(Qf, Kf, Vf, cosb, sinb, Qr, Kr, Vr);

  // 4) local window attention (overwrites Qf region as Lout)
  local_attn<<<dim3(S_/WIN_, H_, B_), 256, 0, stream>>>(Qr, Kr, Vr, Lout);

  // 5) linear attention + hybrid combine
  linear_attn<<<B_*H_, 256, 0, stream>>>(Qr, Kr, Vr, Lout, Hyb);

  // 6) output projection
  gemm_bf16<<<dim3(HID_/128, BS_/128), 256, 0, stream>>>(Hyb, Wob, out, BS_, HID_, HID_);
}
